// EquivariantMHA_32968168964310
// MI455X (gfx1250) — matrix operand order repaired and checked by reading, not compiled, not run on hardware
//
#include <hip/hip_runtime.h>
#include <hip/hip_bf16.h>
#include <stdint.h>
#include <stddef.h>

// ---------------------------------------------------------------------------
// EquivariantMHA forward for MI455X (gfx1250), wave32, bf16 WMMA pipeline.
//   1) cvt x -> bf16; transpose-cvt Wq/Wk/Wv/Wo to [N,K] bf16
//   2) bias[h,q,k] = sum_c ra[c,h]*Q_basis[q,k,c]  -> bf16 [H,S,S] (one pass
//      over the 134 MB Q_basis tensor; dominant memory term)
//   3) GEMM (WMMA bf16) x*W -> fused epilogue: +bias, LayerNorm(q,k) over
//      HD=64 via xor-shuffle row reductions, pack q/k [B,H,S,HD] bf16
//      (q pre-scaled by 1/sqrt(64)), v transposed [B,H,HD,S] bf16
//   4) flash attention: QK^T WMMA; bias tiles streamed into LDS by the
//      Tensor Data Mover (tensor_load_to_lds + s_wait_tensorcnt), double
//      buffered against compute; online softmax; PV WMMA with P transposed
//      through a wave-private LDS strip
//   5) GEMM (WMMA bf16) ctx*Wo + bo -> f32 output
// ---------------------------------------------------------------------------

typedef __attribute__((ext_vector_type(16))) __bf16 v16bf;
typedef __attribute__((ext_vector_type(8)))  __bf16 v8bf;
typedef __attribute__((ext_vector_type(8)))  float  v8f;
typedef __attribute__((ext_vector_type(4)))  unsigned int u32x4;
typedef __attribute__((ext_vector_type(8)))  int i32x8;
typedef __attribute__((ext_vector_type(4)))  int i32x4;

#if defined(__has_builtin)
#if __has_builtin(__builtin_amdgcn_tensor_load_to_lds) && \
    __has_builtin(__builtin_amdgcn_s_wait_tensorcnt)
#define USE_TDM 1
#endif
#endif
#ifndef USE_TDM
#define USE_TDM 0
#endif

static constexpr int B_  = 2;
static constexpr int S_  = 2048;
static constexpr int D_  = 1024;
static constexpr int H_  = 16;
static constexpr int HD_ = 64;
static constexpr int C_  = 8;
static constexpr int M_  = B_ * S_;   // 4096 rows (b,s)
static constexpr int N_  = H_ * HD_;  // 1024
static constexpr int K_  = D_;        // 1024
static constexpr float EPS_ = 1e-6f;

__device__ __forceinline__ unsigned short f2bf(float f) {
  union { float f; unsigned u; } a; a.f = f;
  unsigned r = a.u + 0x7FFFu + ((a.u >> 16) & 1u);   // round-nearest-even
  return (unsigned short)(r >> 16);
}
__device__ __forceinline__ float bf2f(unsigned short b) {
  union { unsigned u; float f; } a; a.u = ((unsigned)b) << 16;
  return a.f;
}

// A fragment (16x32 MxK, bf16): lanes 0-15 M=lane K{0-7,16-23}, lanes 16-31
// same M, K{8-15,24-31}. Both halves are two contiguous 16B reads.
__device__ __forceinline__ v16bf load_frag_a(const unsigned short* base,
                                             int stride, int kbase) {
  int lane = threadIdx.x & 31;
  int m = lane & 15, lh = lane >> 4;
  const unsigned short* p = base + (size_t)m * stride + kbase + lh * 8;
  union { v16bf v; v8bf h2[2]; } u;
  u.h2[0] = *(const v8bf*)(p);        // K = kbase + lh*8 .. +7
  u.h2[1] = *(const v8bf*)(p + 16);   // K = kbase + 16 + lh*8 .. +7
  return u.v;
}

// B fragment (32x16 KxN, bf16) loaded from B^T stored [N,K] row-major:
// lane n = column, lanes 0-15 K{0-7,16-23}, lanes 16-31 K{8-15,24-31}.
__device__ __forceinline__ v16bf load_frag_b(const unsigned short* baseT,
                                             int stride, int kbase) {
  int lane = threadIdx.x & 31;
  int n = lane & 15, lh = lane >> 4;
  const unsigned short* p = baseT + (size_t)n * stride + kbase + lh * 8;
  union { v16bf v; v8bf h2[2]; } u;
  u.h2[0] = *(const v8bf*)(p);
  u.h2[1] = *(const v8bf*)(p + 16);
  return u.v;
}

__device__ __forceinline__ v8f wmma_bf16(v16bf a, v16bf b, v8f c) {
  return __builtin_amdgcn_wmma_f32_16x16x32_bf16(false, a, false, b,
                                                 (short)0, c, false, false);
}

#if USE_TDM
// Generic pointers into LDS map to the LDS aperture with addr[31:0] = LDS
// byte offset (ISA 10.2), which is what D#.lds_addr wants.
__device__ __forceinline__ unsigned lds_byte_off(const void* p) {
  return (unsigned)(uintptr_t)p;
}

// TDM: DMA a 2D bf16 tile [rows x cols], row stride `strideElems`, from
// global memory into LDS at byte offset ldsOff. Tracked by TENSORcnt.
// D# layout per CDNA5 ISA (group0 128b, group1 256b).
__device__ __forceinline__ void tdm_load_tile_2d(const unsigned short* gsrc,
                                                 unsigned ldsOff,
                                                 unsigned cols, unsigned rows,
                                                 unsigned strideElems) {
  unsigned long long ga = (unsigned long long)(uintptr_t)gsrc;  // byte addr
  u32x4 g0;
  g0[0] = 1u;                                   // count=1, user descriptor
  g0[1] = ldsOff;                               // lds_addr (bytes)
  g0[2] = (unsigned)(ga & 0xFFFFFFFFu);         // global_addr[31:0]
  g0[3] = (unsigned)((ga >> 32) & 0x01FFFFFFu)  // global_addr[56:32]
          | (2u << 30);                         // type=2 ("image")
  const unsigned TD = 1u << 30;                 // huge tensor dims: no OOB clip
  i32x8 g1;
  g1[0] = (int)(1u << 16);                      // wg_mask=0, data_size=1 (2B)
  g1[1] = (int)((TD & 0xFFFFu) << 16);          // tensor_dim0[15:0]
  g1[2] = (int)(((TD >> 16) & 0xFFFFu)          // tensor_dim0[31:16]
                | ((TD & 0xFFFFu) << 16));      // tensor_dim1[15:0]
  g1[3] = (int)(((TD >> 16) & 0xFFFFu)          // tensor_dim1[31:16]
                | (cols << 16));                // tile_dim0
  g1[4] = (int)(rows);                          // tile_dim1 (tile_dim2=0)
  g1[5] = (int)strideElems;                     // tensor_dim0_stride[31:0]
  g1[6] = 0;                                    // stride hi / dim1_stride lo
  g1[7] = 0;
  i32x4 z4 = {0, 0, 0, 0};
#if __clang_major__ >= 23
  i32x8 z8 = {0, 0, 0, 0, 0, 0, 0, 0};
  __builtin_amdgcn_tensor_load_to_lds(g0, g1, z4, z4, z8, 0);
#else
  __builtin_amdgcn_tensor_load_to_lds(g0, g1, z4, z4, 0);
#endif
}
#endif  // USE_TDM

// ---------------------------------------------------------------------------
__global__ void cvt_bf16_kernel(const float* __restrict__ in,
                                unsigned short* __restrict__ out, int n) {
  int i = blockIdx.x * blockDim.x + threadIdx.x;
  if (i < n) out[i] = f2bf(in[i]);
}

// in: [rows, cols] f32 row-major -> out: [cols, rows] bf16 row-major
__global__ void transpose_bf16_kernel(const float* __restrict__ in,
                                      unsigned short* __restrict__ out,
                                      int rows, int cols) {
  int i = blockIdx.x * blockDim.x + threadIdx.x;
  if (i < rows * cols) {
    int r = i / cols, c = i - r * cols;
    out[(size_t)c * rows + r] = f2bf(in[i]);
  }
}

// ---------------------------------------------------------------------------
// bias[h][q][k] = sum_c ra[c][h] * Qb[q][k][c]   (bf16 output)
// One 32B contiguous read of Q_basis per thread; writes coalesced over k.
__global__ __launch_bounds__(256)
void bias_kernel(const float* __restrict__ Qb, const float* __restrict__ RA,
                 unsigned short* __restrict__ BiasOut) {
  __shared__ float ra[C_ * H_];
  int t = threadIdx.x;
  if (t < C_ * H_) ra[t] = RA[t];
  __syncthreads();
  int q = blockIdx.y;
  int k = blockIdx.x * 256 + t;
  const float* p = Qb + ((size_t)q * S_ + k) * C_;
  __builtin_prefetch(p + 256 * C_, 0, 1);   // global_prefetch_b8: stream ahead
  float v[C_];
#pragma unroll
  for (int c = 0; c < C_; ++c) v[c] = p[c];
#pragma unroll
  for (int hh = 0; hh < H_; ++hh) {
    float acc = 0.f;
#pragma unroll
    for (int c = 0; c < C_; ++c) acc = fmaf(v[c], ra[c * H_ + hh], acc);
    BiasOut[((size_t)hh * S_ + q) * S_ + k] = f2bf(acc);
  }
}

// ---------------------------------------------------------------------------
// WMMA GEMM: A[M,K] bf16 x Bt[N,K] bf16 -> epilogue by mode.
// Block 256 thr = 8 waves, block tile 128x128, wave tile 32x64 (head-aligned).
// mode 0: out f32 [M,N] (+bias)            -> final output projection
// mode 1: +bias, LayerNorm over HD, *extraScale, bf16 [B,H,S,HD] (q/k)
// mode 2: +bias, bf16 transposed [B,H,HD,S]                     (v)
__global__ __launch_bounds__(256)
void gemm_wmma_kernel(const unsigned short* __restrict__ A,
                      const unsigned short* __restrict__ Bt,
                      const float* __restrict__ bias,
                      float* __restrict__ outF,
                      unsigned short* __restrict__ outB,
                      const float* __restrict__ gamma,
                      float extraScale, int mode) {
  int w = threadIdx.x >> 5;
  int lane = threadIdx.x & 31;
  int lh = lane >> 4, nloc = lane & 15;
  int wn = w & 1, wm = w >> 1;
  int rowBase = blockIdx.y * 128 + wm * 32;
  int colBase = blockIdx.x * 128 + wn * 64;   // 64-wide => one head per wave

  const v8f zf = {0.f, 0.f, 0.f, 0.f, 0.f, 0.f, 0.f, 0.f};
  v8f acc[8];
#pragma unroll
  for (int i = 0; i < 8; ++i) acc[i] = zf;

  const unsigned short* Arow0 = A + (size_t)rowBase * K_;
  const unsigned short* Arow1 = A + (size_t)(rowBase + 16) * K_;

  for (int k = 0; k < K_; k += 32) {
    v16bf a0 = load_frag_a(Arow0, K_, k);
    v16bf a1 = load_frag_a(Arow1, K_, k);
#pragma unroll
    for (int nj = 0; nj < 4; ++nj) {
      v16bf b = load_frag_b(Bt + (size_t)(colBase + nj * 16) * K_, K_, k);
      acc[nj]     = wmma_bf16(a0, b, acc[nj]);
      acc[4 + nj] = wmma_bf16(a1, b, acc[4 + nj]);
    }
  }

  // accumulator layout: VGPR r -> row (lh*8 + r), lane%16 -> col within tile
#pragma unroll
  for (int mi = 0; mi < 2; ++mi) {
#pragma unroll
    for (int nj = 0; nj < 4; ++nj) {
      float bcol = bias[colBase + nj * 16 + nloc];
#pragma unroll
      for (int r = 0; r < 8; ++r) acc[mi * 4 + nj][r] += bcol;
    }
    if (mode == 0) {
#pragma unroll
      for (int nj = 0; nj < 4; ++nj) {
        int col = colBase + nj * 16 + nloc;
#pragma unroll
        for (int r = 0; r < 8; ++r) {
          int row = rowBase + mi * 16 + lh * 8 + r;
          outF[(size_t)row * N_ + col] = acc[mi * 4 + nj][r];
        }
      }
    } else {
      int hd = colBase >> 6;   // head index (wave col span head-aligned)
#pragma unroll
      for (int r = 0; r < 8; ++r) {
        int row = rowBase + mi * 16 + lh * 8 + r;
        int b = row >> 11, s = row & (S_ - 1);
        if (mode == 1) {
          // LayerNorm over the 64-wide head span: 4 tiles x 16 lanes
          float sum = 0.f;
#pragma unroll
          for (int nj = 0; nj < 4; ++nj) sum += acc[mi * 4 + nj][r];
#pragma unroll
          for (int msk = 1; msk <= 8; msk <<= 1) sum += __shfl_xor(sum, msk);
          float mean = sum * (1.f / 64.f);
          float vsum = 0.f;
#pragma unroll
          for (int nj = 0; nj < 4; ++nj) {
            float d = acc[mi * 4 + nj][r] - mean;
            vsum += d * d;
          }
#pragma unroll
          for (int msk = 1; msk <= 8; msk <<= 1) vsum += __shfl_xor(vsum, msk);
          float rstd = __frsqrt_rn(vsum * (1.f / 64.f) + EPS_);
#pragma unroll
          for (int nj = 0; nj < 4; ++nj) {
            int d = nj * 16 + nloc;
            float y = (acc[mi * 4 + nj][r] - mean) * rstd * gamma[d] * extraScale;
            outB[(((size_t)b * H_ + hd) * S_ + s) * HD_ + d] = f2bf(y);
          }
        } else {  // mode 2: V, stored transposed [B,H,HD,S]
#pragma unroll
          for (int nj = 0; nj < 4; ++nj) {
            int d = nj * 16 + nloc;
            outB[(((size_t)b * H_ + hd) * HD_ + d) * S_ + s] =
                f2bf(acc[mi * 4 + nj][r]);
          }
        }
      }
    }
  }
}

// ---------------------------------------------------------------------------
// Flash attention, 4 waves/block, 16 q-rows per wave, 64-wide kv steps.
// Q,K: [B,H,S,HD] bf16 (q pre-scaled by 1/8). V: [B,H,HD,S] bf16 (transposed).
// Bias: [H,S,S] bf16 -> streamed into LDS by the TDM, double buffered.
// Ctx out: [B,S,H,HD] bf16.
__global__ __launch_bounds__(128)
void flash_attn_kernel(const unsigned short* __restrict__ Q,
                       const unsigned short* __restrict__ Kc,
                       const unsigned short* __restrict__ Vt,
                       const unsigned short* __restrict__ Bias,
                       unsigned short* __restrict__ Ctx) {
  __shared__ __align__(16) unsigned short Plds[4][16 * 64];    // 8 KB
#if USE_TDM
  __shared__ __align__(16) unsigned short BiasLds[2][64 * 64]; // 16 KB
#endif
  int w = threadIdx.x >> 5, lane = threadIdx.x & 31;
  int lh = lane >> 4, nloc = lane & 15;
  int bh = blockIdx.y;
  int b = bh / H_, hd = bh - b * H_;
  int qBlock = blockIdx.x * 64;
  int qTile = qBlock + w * 16;

  const unsigned short* qP = Q + ((size_t)bh * S_ + qTile) * HD_;
  const unsigned short* kP = Kc + (size_t)bh * S_ * HD_;
  const unsigned short* vP = Vt + (size_t)bh * HD_ * S_;
  const unsigned short* biasBlkP = Bias + ((size_t)hd * S_ + qBlock) * S_;

  v16bf aq0 = load_frag_a(qP, HD_, 0);    // K(=HD) 0..31
  v16bf aq1 = load_frag_a(qP, HD_, 32);   // K(=HD) 32..63

  const v8f zf = {0.f, 0.f, 0.f, 0.f, 0.f, 0.f, 0.f, 0.f};
  v8f O[4];
  float mrow[8], lrow[8];
#pragma unroll
  for (int t = 0; t < 4; ++t) O[t] = zf;
#pragma unroll
  for (int r = 0; r < 8; ++r) { mrow[r] = -1e30f; lrow[r] = 0.f; }

#if USE_TDM
  // Prologue: wave 0 DMAs the first 64x64 bias tile into buffer 0.
  if (w == 0)
    tdm_load_tile_2d(biasBlkP, lds_byte_off(&BiasLds[0][0]), 64, 64, S_);
#endif

  for (int kb = 0; kb < S_; kb += 64) {
    // ---- S = q * k^T  (independent of bias -> overlaps the TDM) ----
    v8f sAcc[4];
#pragma unroll
    for (int t = 0; t < 4; ++t) {
      sAcc[t] = zf;
      const unsigned short* kTile = kP + (size_t)(kb + t * 16) * HD_;
      sAcc[t] = wmma_bf16(aq0, load_frag_b(kTile, HD_, 0), sAcc[t]);
      sAcc[t] = wmma_bf16(aq1, load_frag_b(kTile, HD_, 32), sAcc[t]);
    }
#if USE_TDM
    int cur = (kb >> 6) & 1;
    if (w == 0) __builtin_amdgcn_s_wait_tensorcnt(0);  // tile `cur` landed
    __syncthreads();  // publish tile; also gates overwrite of old buffer
    if (w == 0 && kb + 64 < S_)  // kick next tile while we compute
      tdm_load_tile_2d(biasBlkP + (kb + 64),
                       lds_byte_off(&BiasLds[cur ^ 1][0]), 64, 64, S_);
    // ---- add relative-position bias from LDS (ds_load_u16) ----
#pragma unroll
    for (int t = 0; t < 4; ++t) {
#pragma unroll
      for (int r = 0; r < 8; ++r) {
        int qr = w * 16 + lh * 8 + r;
        sAcc[t][r] += bf2f(BiasLds[cur][qr * 64 + t * 16 + nloc]);
      }
    }
#else
    // Fallback: scattered global bf16 loads
#pragma unroll
    for (int t = 0; t < 4; ++t) {
#pragma unroll
      for (int r = 0; r < 8; ++r) {
        int qr = w * 16 + lh * 8 + r;
        sAcc[t][r] += bf2f(biasBlkP[(size_t)qr * S_ + kb + t * 16 + nloc]);
      }
    }
#endif
    // ---- online softmax (row = (lane-half, vgpr r); reduce 16-lane group) --
#pragma unroll
    for (int r = 0; r < 8; ++r) {
      float rm = fmaxf(fmaxf(sAcc[0][r], sAcc[1][r]),
                       fmaxf(sAcc[2][r], sAcc[3][r]));
#pragma unroll
      for (int msk = 1; msk <= 8; msk <<= 1) rm = fmaxf(rm, __shfl_xor(rm, msk));
      float mn = fmaxf(mrow[r], rm);
      float corr = __expf(mrow[r] - mn);
      float rs = 0.f;
#pragma unroll
      for (int t = 0; t < 4; ++t) {
        float p = __expf(sAcc[t][r] - mn);
        sAcc[t][r] = p;
        rs += p;
      }
#pragma unroll
      for (int msk = 1; msk <= 8; msk <<= 1) rs += __shfl_xor(rs, msk);
      lrow[r] = lrow[r] * corr + rs;
      mrow[r] = mn;
#pragma unroll
      for (int t = 0; t < 4; ++t) O[t][r] *= corr;
    }
    // ---- transpose P via wave-private LDS strip (DS in-order per wave) ----
#pragma unroll
    for (int t = 0; t < 4; ++t)
#pragma unroll
      for (int r = 0; r < 8; ++r)
        Plds[w][(lh * 8 + r) * 64 + t * 16 + nloc] = f2bf(sAcc[t][r]);
    v16bf ap0 = load_frag_a(&Plds[w][0], 64, 0);
    v16bf ap1 = load_frag_a(&Plds[w][0], 64, 32);
    // ---- O += P * V  (V^T rows are contiguous over kv) ----
#pragma unroll
    for (int t = 0; t < 4; ++t) {
      const unsigned short* vTile = vP + (size_t)(t * 16) * S_;
      O[t] = wmma_bf16(ap0, load_frag_b(vTile, S_, kb), O[t]);
      O[t] = wmma_bf16(ap1, load_frag_b(vTile, S_, kb + 32), O[t]);
    }
  }
  // ---- finalize: O /= l, write ctx [B,S,H,HD] bf16 ----
#pragma unroll
  for (int r = 0; r < 8; ++r) {
    float inv = 1.f / lrow[r];
    int q = qTile + lh * 8 + r;
#pragma unroll
    for (int t = 0; t < 4; ++t) {
      int d = t * 16 + nloc;
      Ctx[(((size_t)b * S_ + q) * H_ + hd) * HD_ + d] = f2bf(O[t][r] * inv);
    }
  }
}

// ---------------------------------------------------------------------------
extern "C" void kernel_launch(void* const* d_in, const int* in_sizes, int n_in,
                              void* d_out, int out_size, void* d_ws,
                              size_t ws_size, hipStream_t stream) {
  const float* x   = (const float*)d_in[0];
  const float* Qb  = (const float*)d_in[1];
  const float* Wq  = (const float*)d_in[2];
  const float* bq  = (const float*)d_in[3];
  const float* Wk  = (const float*)d_in[4];
  const float* bk  = (const float*)d_in[5];
  const float* Wv  = (const float*)d_in[6];
  const float* bv  = (const float*)d_in[7];
  const float* qls = (const float*)d_in[8];
  const float* kls = (const float*)d_in[9];
  const float* ra  = (const float*)d_in[10];
  const float* Wo  = (const float*)d_in[11];
  const float* bo  = (const float*)d_in[12];
  float* out = (float*)d_out;

  // workspace carve-up (all bf16 stored as raw u16 bits)
  char* p = (char*)d_ws;
  size_t off = 0;
  auto take = [&](size_t bytes) {
    char* r = p + off;
    off += (bytes + 255) & ~size_t(255);
    return r;
  };
  unsigned short* xb   = (unsigned short*)take((size_t)M_ * K_ * 2);   // 8 MB
  unsigned short* wqT  = (unsigned short*)take((size_t)N_ * K_ * 2);   // 2 MB
  unsigned short* wkT  = (unsigned short*)take((size_t)N_ * K_ * 2);
  unsigned short* wvT  = (unsigned short*)take((size_t)N_ * K_ * 2);
  unsigned short* woT  = (unsigned short*)take((size_t)N_ * K_ * 2);
  unsigned short* qb2  = (unsigned short*)take((size_t)B_ * H_ * S_ * HD_ * 2);
  unsigned short* kb2  = (unsigned short*)take((size_t)B_ * H_ * S_ * HD_ * 2);
  unsigned short* vT   = (unsigned short*)take((size_t)B_ * H_ * HD_ * S_ * 2);
  unsigned short* ctx  = (unsigned short*)take((size_t)B_ * S_ * H_ * HD_ * 2);
  unsigned short* bias = (unsigned short*)take((size_t)H_ * S_ * S_ * 2); // 134MB
  (void)ws_size; (void)in_sizes; (void)n_in; (void)out_size;

  // 1) input / weight precision + layout prep
  cvt_bf16_kernel<<<(M_ * K_ + 255) / 256, 256, 0, stream>>>(x, xb, M_ * K_);
  transpose_bf16_kernel<<<(K_ * N_ + 255) / 256, 256, 0, stream>>>(Wq, wqT, K_, N_);
  transpose_bf16_kernel<<<(K_ * N_ + 255) / 256, 256, 0, stream>>>(Wk, wkT, K_, N_);
  transpose_bf16_kernel<<<(K_ * N_ + 255) / 256, 256, 0, stream>>>(Wv, wvT, K_, N_);
  transpose_bf16_kernel<<<(K_ * N_ + 255) / 256, 256, 0, stream>>>(Wo, woT, K_, N_);

  // 2) relative-position bias (single pass over Q_basis)
  bias_kernel<<<dim3(S_ / 256, S_), 256, 0, stream>>>(Qb, ra, bias);

  // 3) QKV projections with fused LN / pack epilogues
  dim3 gGrid(N_ / 128, M_ / 128);
  gemm_wmma_kernel<<<gGrid, 256, 0, stream>>>(xb, wqT, bq, nullptr, qb2, qls,
                                              0.125f /* 1/sqrt(HD) */, 1);
  gemm_wmma_kernel<<<gGrid, 256, 0, stream>>>(xb, wkT, bk, nullptr, kb2, kls,
                                              1.0f, 1);
  gemm_wmma_kernel<<<gGrid, 256, 0, stream>>>(xb, wvT, bv, nullptr, vT,
                                              nullptr, 1.0f, 2);

  // 4) flash attention (TDM-streamed bias tiles)
  flash_attn_kernel<<<dim3(S_ / 64, B_ * H_), 128, 0, stream>>>(qb2, kb2, vT,
                                                                bias, ctx);

  // 5) output projection -> f32 result
  gemm_wmma_kernel<<<gGrid, 256, 0, stream>>>(ctx, woT, bo, out, nullptr,
                                              nullptr, 1.0f, 0);
}
